// Embedder_69922067579082
// MI455X (gfx1250) — matrix operand order repaired and checked by reading, not compiled, not run on hardware
//
#include <hip/hip_runtime.h>
#include <hip/hip_bf16.h>
#include <math.h>

#define TSEQ 2048
#define EDIM 64
#define HDIM 64
#define GDIM 192          // 3*H
#define VOCAB 50257
#define NTILES 3142       // ceil(VOCAB/16)
#define MTILES 128        // TSEQ/16

typedef __attribute__((ext_vector_type(16))) _Float16 v16h;
typedef __attribute__((ext_vector_type(8)))  float    v8f;

__device__ __forceinline__ float sigmoidf_(float x) {
    return 1.0f / (1.0f + expf(-x));
}

// ---------------------------------------------------------------------------
// Kernel A: embedding gather fused with input-side gate GEMM for the encoder.
// Gi[t, 0:192] = Wih_enc @ emb[data[t]] + bih_enc   (fully parallel over t)
// ---------------------------------------------------------------------------
__global__ void gi_enc_kernel(const int* __restrict__ data,
                              const float* __restrict__ emb,
                              const float* __restrict__ Wih,
                              const float* __restrict__ bih,
                              float* __restrict__ Gi) {
    __shared__ __align__(16) float xs[EDIM];
    const int t = blockIdx.x;
    const int j = threadIdx.x;           // 0..191
    const int tok = data[t];
    if (j < EDIM) xs[j] = emb[(size_t)tok * EDIM + j];
    __syncthreads();
    float acc = bih[j];
    const float* w = Wih + j * EDIM;
    const float4* x4 = (const float4*)xs;
#pragma unroll
    for (int e4 = 0; e4 < EDIM / 4; ++e4) {
        float4 xv = x4[e4];
        acc = fmaf(w[4 * e4 + 0], xv.x, acc);
        acc = fmaf(w[4 * e4 + 1], xv.y, acc);
        acc = fmaf(w[4 * e4 + 2], xv.z, acc);
        acc = fmaf(w[4 * e4 + 3], xv.w, acc);
    }
    Gi[t * GDIM + j] = acc;
}

// ---------------------------------------------------------------------------
// Kernel B: encoder recurrence. One workgroup, 192 threads (6 waves).
// Weight row lives in VGPRs; hidden state broadcast via LDS.
// ---------------------------------------------------------------------------
__global__ void __launch_bounds__(192) enc_rnn_kernel(const float* __restrict__ Whh,
                                                      const float* __restrict__ bhh,
                                                      const float* __restrict__ Gi,
                                                      float* __restrict__ h_out) {
    __shared__ __align__(16) float hs[HDIM];
    __shared__ float gh[GDIM];
    const int j = threadIdx.x;
    float w[HDIM];
#pragma unroll
    for (int e = 0; e < HDIM; ++e) w[e] = Whh[j * HDIM + e];
    const float bj = bhh[j];
    if (j < HDIM) hs[j] = 0.0f;
    __syncthreads();
    for (int t = 0; t < TSEQ; ++t) {
        float acc = bj;
        const float4* h4 = (const float4*)hs;
#pragma unroll
        for (int e4 = 0; e4 < HDIM / 4; ++e4) {
            float4 hv = h4[e4];
            acc = fmaf(w[4 * e4 + 0], hv.x, acc);
            acc = fmaf(w[4 * e4 + 1], hv.y, acc);
            acc = fmaf(w[4 * e4 + 2], hv.z, acc);
            acc = fmaf(w[4 * e4 + 3], hv.w, acc);
        }
        gh[j] = acc;
        __syncthreads();
        if (j < HDIM) {
            const float* gi = Gi + t * GDIM;
            float r = sigmoidf_(gi[j]            + gh[j]);
            float z = sigmoidf_(gi[HDIM + j]     + gh[HDIM + j]);
            float n = tanhf    (gi[2 * HDIM + j] + r * gh[2 * HDIM + j]);
            hs[j] = (1.0f - z) * n + z * hs[j];
        }
        __syncthreads();
    }
    if (j < HDIM) h_out[j] = hs[j];
}

// ---------------------------------------------------------------------------
// Kernel C: decoder recurrence. One workgroup, 192 threads. Emits hidden
// states DIRECTLY in WMMA f16 A-fragment layout (ISA 7.12.2, 16-bit A 16x32):
//   lane = half*16 + m,   half=(kk/8)&1,  elem jj = (kk&7) + 8*(kk>>4)
// ---------------------------------------------------------------------------
__global__ void __launch_bounds__(192) dec_rnn_kernel(const float* __restrict__ Wih,
                                                      const float* __restrict__ Whh,
                                                      const float* __restrict__ bih,
                                                      const float* __restrict__ bhh,
                                                      const float* __restrict__ h0,
                                                      _Float16* __restrict__ wsA) {
    __shared__ __align__(16) float inp[HDIM];
    __shared__ __align__(16) float hs[HDIM];
    __shared__ float gi[GDIM];
    __shared__ float gh[GDIM];
    const int j = threadIdx.x;
    float wi[HDIM], wh[HDIM];
#pragma unroll
    for (int e = 0; e < HDIM; ++e) wi[e] = Wih[j * HDIM + e];
#pragma unroll
    for (int e = 0; e < HDIM; ++e) wh[e] = Whh[j * HDIM + e];
    const float bi = bih[j], bh = bhh[j];
    if (j < HDIM) { inp[j] = 0.0f; hs[j] = h0[j]; }
    __syncthreads();
    for (int t = 0; t < TSEQ; ++t) {
        float ai = bi, ah = bh;
        const float4* i4 = (const float4*)inp;
        const float4* h4 = (const float4*)hs;
#pragma unroll
        for (int e4 = 0; e4 < HDIM / 4; ++e4) {
            float4 iv = i4[e4];
            float4 hv = h4[e4];
            ai = fmaf(wi[4 * e4 + 0], iv.x, ai); ah = fmaf(wh[4 * e4 + 0], hv.x, ah);
            ai = fmaf(wi[4 * e4 + 1], iv.y, ai); ah = fmaf(wh[4 * e4 + 1], hv.y, ah);
            ai = fmaf(wi[4 * e4 + 2], iv.z, ai); ah = fmaf(wh[4 * e4 + 2], hv.z, ah);
            ai = fmaf(wi[4 * e4 + 3], iv.w, ai); ah = fmaf(wh[4 * e4 + 3], hv.w, ah);
        }
        gi[j] = ai;
        gh[j] = ah;
        __syncthreads();
        if (j < HDIM) {
            float r = sigmoidf_(gi[j]            + gh[j]);
            float z = sigmoidf_(gi[HDIM + j]     + gh[HDIM + j]);
            float n = tanhf    (gi[2 * HDIM + j] + r * gh[2 * HDIM + j]);
            float out = (1.0f - z) * n + z * hs[j];
            hs[j]  = out;
            inp[j] = out;
            // scatter into WMMA A-fragment layout: tile mt=t/16, row m=t%16, k=j
            int mt = t >> 4, m = t & 15;
            int ks = j >> 5, kk = j & 31;
            int half = (kk >> 3) & 1;
            int jj   = (kk & 7) + 8 * (kk >> 4);
            int lane = half * 16 + m;
            wsA[(((mt * 2 + ks) * 32 + lane) << 4) + jj] = (_Float16)out;
        }
        __syncthreads();
    }
}

// ---------------------------------------------------------------------------
// Kernel D: convert/pack W_out (fp32 row-major [V,64]) into f16 WMMA
// B-fragments: lane n=L%16 holds K = (jj&7) + 8*(L/16) + 16*(jj>>3) of column (nt*16 + n).
// ---------------------------------------------------------------------------
__global__ void packB_kernel(const float* __restrict__ W, _Float16* __restrict__ wsB) {
    int idx = blockIdx.x * 256 + threadIdx.x;
    const int total = NTILES * 2 * 32 * 16;
    if (idx >= total) return;
    int jj   = idx & 15;
    int lane = (idx >> 4) & 31;
    int ks   = (idx >> 9) & 1;
    int nt   = idx >> 10;
    int row  = nt * 16 + (lane & 15);
    int col  = ks * 32 + (jj & 7) + 8 * (lane >> 4) + 16 * (jj >> 3);
    float v = (row < VOCAB) ? W[(size_t)row * HDIM + col] : 0.0f;
    wsB[idx] = (_Float16)v;
}

// ---------------------------------------------------------------------------
// Kernel E: logits = outs @ W_out^T + b_out via v_wmma_f32_16x16x32_f16.
// 8 waves/block; each wave: one M-tile x 4 N-tiles, K=64 (2 wmma k-steps),
// A fragment reused across the 4 N-tiles. Store-bandwidth bound (412 MB).
// Wave id is forced into an SGPR so tile guards are scalar branches.
// ---------------------------------------------------------------------------
__global__ void __launch_bounds__(256) out_gemm_kernel(const _Float16* __restrict__ wsA,
                                                       const _Float16* __restrict__ wsB,
                                                       const float* __restrict__ b_out,
                                                       float* __restrict__ logits) {
    const int lane = threadIdx.x & 31;
    // Uniform per wave -> pin it in an SGPR so nt guards become s_cmp/s_cbranch.
    const int wave = __builtin_amdgcn_readfirstlane(threadIdx.x >> 5);
    const int mt = blockIdx.y;
    const int ntBase = (blockIdx.x * 8 + wave) * 4;
    if (ntBase >= NTILES) return;

    const v16h* Af = (const v16h*)wsA;
    const v16h* Bf = (const v16h*)wsB;

    v16h a0 = Af[(mt * 2 + 0) * 32 + lane];
    v16h a1 = Af[(mt * 2 + 1) * 32 + lane];

    v8f acc[4] = {};
#pragma unroll
    for (int i = 0; i < 4; ++i) {
        int nt = ntBase + i;
        if (nt < NTILES) {
            const v16h* bp = &Bf[(nt * 2) * 32 + lane];
            __builtin_prefetch((const void*)(bp + 64), 0, 3);  // prime near cache for tile i+1
            v16h b0 = bp[0];
            v16h b1 = bp[32];
            acc[i] = __builtin_amdgcn_wmma_f32_16x16x32_f16(false, a0, false, b0,
                                                            (short)0, acc[i], false, false);
            acc[i] = __builtin_amdgcn_wmma_f32_16x16x32_f16(false, a1, false, b1,
                                                            (short)0, acc[i], false, false);
        }
    }

    const int rowBase = mt * 16 + (lane >> 4) * 8;   // C/D layout: vgpr r -> M = r + 8*(lane/16)
#pragma unroll
    for (int i = 0; i < 4; ++i) {
        int nt = ntBase + i;
        if (nt >= NTILES) continue;
        int col = nt * 16 + (lane & 15);
        if (col >= VOCAB) continue;
        float bv = b_out[col];
        float* outp = logits + (size_t)rowBase * VOCAB + col;
#pragma unroll
        for (int r = 0; r < 8; ++r) {
            outp[(size_t)r * VOCAB] = acc[i][r] + bv;
        }
    }
}

// ---------------------------------------------------------------------------
extern "C" void kernel_launch(void* const* d_in, const int* in_sizes, int n_in,
                              void* d_out, int out_size, void* d_ws, size_t ws_size,
                              hipStream_t stream) {
    const int*   data    = (const int*)  d_in[0];
    // d_in[1] = Type: Type[0,0]==1 by construction -> single (map) branch
    const float* emb     = (const float*)d_in[2];
    const float* Wih_enc = (const float*)d_in[3];
    const float* Whh_enc = (const float*)d_in[4];
    const float* bih_enc = (const float*)d_in[5];
    const float* bhh_enc = (const float*)d_in[6];
    const float* Wih_dec = (const float*)d_in[7];
    const float* Whh_dec = (const float*)d_in[8];
    const float* bih_dec = (const float*)d_in[9];
    const float* bhh_dec = (const float*)d_in[10];
    const float* W_out   = (const float*)d_in[11];
    const float* b_out   = (const float*)d_in[12];
    float* logits = (float*)d_out;

    // Workspace carve-up (≈ 8.3 MB total)
    char* ws = (char*)d_ws;
    const size_t wsA_bytes = (size_t)MTILES * 2 * 32 * 16 * sizeof(_Float16);   // 256 KB
    const size_t wsB_bytes = (size_t)NTILES * 2 * 32 * 16 * sizeof(_Float16);   // ~6.4 MB
    const size_t gi_bytes  = (size_t)TSEQ * GDIM * sizeof(float);               // 1.5 MB
    _Float16* wsA  = (_Float16*)(ws);
    _Float16* wsB  = (_Float16*)(ws + wsA_bytes);
    float*    Gi   = (float*)   (ws + wsA_bytes + wsB_bytes);
    float*    hEnc = (float*)   (ws + wsA_bytes + wsB_bytes + gi_bytes);

    // A: embed + encoder input-gate GEMM (parallel over all 2048 steps)
    gi_enc_kernel<<<TSEQ, GDIM, 0, stream>>>(data, emb, Wih_enc, bih_enc, Gi);
    // D: pack W_out into WMMA B fragments (independent of A/B/C)
    {
        const int total = NTILES * 2 * 32 * 16;
        packB_kernel<<<(total + 255) / 256, 256, 0, stream>>>(W_out, wsB);
    }
    // B: encoder recurrence (sequential, single WG)
    enc_rnn_kernel<<<1, GDIM, 0, stream>>>(Whh_enc, bhh_enc, Gi, hEnc);
    // C: decoder recurrence, emits A-fragments (sequential, single WG)
    dec_rnn_kernel<<<1, GDIM, 0, stream>>>(Wih_dec, Whh_dec, bih_dec, bhh_dec, hEnc, wsA);
    // E: vocab projection GEMM with WMMA
    dim3 grid((NTILES + 31) / 32, MTILES);
    out_gemm_kernel<<<grid, 256, 0, stream>>>(wsA, wsB, b_out, logits);
}
